// AverageAttention_20280835572052
// MI455X (gfx1250) — compile-verified, matrix operand order checked by reading, not run on hardware
//
#include <hip/hip_runtime.h>

// Problem constants (B=8, T=2048, D=2048)
#define DD   2048
#define KK   4096
#define MTOT 16384     // B*T
#define CH   16        // scan chunks per column
#define CLEN 128       // T / CH

typedef __attribute__((ext_vector_type(16))) __bf16 v16bf;
typedef __attribute__((ext_vector_type(8)))  float  v8f;

#if defined(__has_builtin)
#if __has_builtin(__builtin_amdgcn_cvt_pk_bf16_f32)
#define HAVE_PK_BF16 1
#endif
#endif

// Pack two f32 -> two bf16 (round-half-up) in 3 VALU: add, add, v_perm_b32.
__device__ __forceinline__ unsigned pack2bf(float a, float b) {
#ifdef HAVE_PK_BF16
    auto r = __builtin_amdgcn_cvt_pk_bf16_f32(a, b);
    unsigned u;
    __builtin_memcpy(&u, &r, 4);
    return u;
#else
    unsigned ua = __float_as_uint(a) + 0x8000u;
    unsigned ub = __float_as_uint(b) + 0x8000u;
    // result = { ub[31:16], ua[31:16] }  -> single v_perm_b32
    return __byte_perm(ua, ub, 0x7632);
#endif
}
__device__ __forceinline__ float sigm(float x) {
    return 1.0f / (1.0f + __expf(-x));
}

// ---------------- Pass 1: per-chunk partial sums over T ----------------
__global__ void __launch_bounds__(256)
avg_partials(const float* __restrict__ x, float* __restrict__ part) {
    int idx = blockIdx.x * 256 + threadIdx.x;       // (b, c, d) flat
    int d = idx & (DD - 1);
    int c = (idx >> 11) & (CH - 1);
    int b = idx >> 15;
    const float* p = x + (size_t)(b * 2048 + c * CLEN) * DD + d;
    float s = 0.f;
#pragma unroll 8
    for (int t = 0; t < CLEN; ++t) s += p[(size_t)t * DD];
    part[idx] = s;                                   // [b][c][d]
}

// ---------------- Pass 2: scan within chunk, divide by row index -------
__global__ void __launch_bounds__(256)
avg_scan(const float* __restrict__ x, const float* __restrict__ part,
         float* __restrict__ avg) {
    int idx = blockIdx.x * 256 + threadIdx.x;
    int d = idx & (DD - 1);
    int c = (idx >> 11) & (CH - 1);
    int b = idx >> 15;
    float acc = 0.f;
    for (int j = 0; j < c; ++j) acc += part[((b * CH + j) << 11) + d];
    size_t base = (size_t)(b * 2048 + c * CLEN) * DD + d;
    const float* p = x + base;
    float* ap = avg + base;
    int t0 = c * CLEN;
#pragma unroll 4
    for (int t = 0; t < CLEN; ++t) {
        acc += p[(size_t)t * DD];
        ap[(size_t)t * DD] = acc / (float)(t0 + t + 1);
    }
}

// ---------------- Fused dual-gate WMMA GEMM + sigmoid combine ----------
// Tile: 128 rows x 64 d-cols, computing gates[:, n] and gates[:, n+2048].
// Double-buffered LDS, one barrier per k-step; rows padded to 40 ushorts
// (80 B = 20 dwords) -> conflict-free ds_load_b128 fragment reads.
__global__ void __launch_bounds__(256)
gated_gemm(const float* __restrict__ x, const float* __restrict__ avg,
           const float* __restrict__ Wg, const float* __restrict__ bg,
           float* __restrict__ out) {
    __shared__ unsigned short sA[2][128 * 40];
    __shared__ unsigned short sB[2][128 * 40];

    const int tid   = threadIdx.x;
    const int lane  = tid & 31;
    const int wave  = tid >> 5;
    const int waveM = wave & 3;               // 4 waves along M
    const int waveN = wave >> 2;              // 2 waves along N
    const int rBase = waveM * 32;
    const int cBase = waveN * 32;
    const int lrow  = lane & 15;
    const int hb    = lane >> 4;              // half-wave selects K subgroup

    const int mBase = blockIdx.y * 128;       // row tile in [0, 16384)
    const int nBase = blockIdx.x * 64;        // d tile in [0, 2048)

    // Precomputed staging slots: 4 float4 per thread per tile
    size_t aOff[4], bOff[4];
    int    ldsOff[4];
#pragma unroll
    for (int i = 0; i < 4; ++i) {
        int q   = tid + i * 256;              // 0..1023 float4 slots
        int row = q >> 3;
        int kq  = q & 7;
        aOff[i]   = (size_t)(mBase + row) * DD + kq * 4;
        int o     = (row < 64) ? (nBase + row) : (DD + nBase + (row - 64));
        bOff[i]   = (size_t)o * KK + kq * 4;
        ldsOff[i] = row * 40 + kq * 4;
    }
    // Fragment base offsets (ushort units) per ISA bf16 A/B layouts
    int aFOff[2], bFOff[2][2];
#pragma unroll
    for (int mi = 0; mi < 2; ++mi)
        aFOff[mi] = (rBase + mi * 16 + lrow) * 40 + hb * 8;
#pragma unroll
    for (int g = 0; g < 2; ++g)
#pragma unroll
        for (int ni = 0; ni < 2; ++ni)
            bFOff[g][ni] = (g * 64 + cBase + ni * 16 + lrow) * 40 + hb * 8;

    v8f acc[2][2][2] = {};                    // [gate][mi][ni]

    // ---- prologue: stage k-step 0 into buffer 0 ----
    {
#pragma unroll
        for (int i = 0; i < 4; ++i) {
            const float4 va = *(const float4*)(x + aOff[i]);   // kb=0 < DD
            const float4 vb = *(const float4*)(Wg + bOff[i]);
            uint2 pa, pb;
            pa.x = pack2bf(va.x, va.y); pa.y = pack2bf(va.z, va.w);
            pb.x = pack2bf(vb.x, vb.y); pb.y = pack2bf(vb.z, vb.w);
            *(uint2*)&sA[0][ldsOff[i]] = pa;
            *(uint2*)&sB[0][ldsOff[i]] = pb;
        }
    }
    __syncthreads();

#pragma unroll 2
    for (int kb = 0; kb < KK; kb += 32) {
        const int cur = (kb >> 5) & 1;
        const int nxt = cur ^ 1;
        const bool hasNext = (kb + 32) < KK;

        // ---- issue next-step global loads early (hide under WMMAs) ----
        float4 ra[4], rb[4];
        if (hasNext) {
            const int kn = kb + 32;
            const float* aSrc = (kn < DD) ? (x + kn) : (avg + (kn - DD));
#pragma unroll
            for (int i = 0; i < 4; ++i) ra[i] = *(const float4*)(aSrc + aOff[i]);
#pragma unroll
            for (int i = 0; i < 4; ++i) rb[i] = *(const float4*)(Wg + bOff[i] + kn);
        }

        // ---- fragment loads + 8 WMMAs on current buffer ----
        union Frag { uint4 u[2]; v16bf v; };
        Frag a[2], bfr[2][2];
#pragma unroll
        for (int mi = 0; mi < 2; ++mi) {
            const unsigned short* p = &sA[cur][aFOff[mi]];
            a[mi].u[0] = *(const uint4*)p;         // K 0..7  / 8..15
            a[mi].u[1] = *(const uint4*)(p + 16);  // K 16..23 / 24..31
        }
#pragma unroll
        for (int g = 0; g < 2; ++g)
#pragma unroll
            for (int ni = 0; ni < 2; ++ni) {
                const unsigned short* p = &sB[cur][bFOff[g][ni]];
                bfr[g][ni].u[0] = *(const uint4*)p;
                bfr[g][ni].u[1] = *(const uint4*)(p + 16);
            }
#pragma unroll
        for (int g = 0; g < 2; ++g)
#pragma unroll
            for (int mi = 0; mi < 2; ++mi)
#pragma unroll
                for (int ni = 0; ni < 2; ++ni)
                    acc[g][mi][ni] = __builtin_amdgcn_wmma_f32_16x16x32_bf16(
                        false, a[mi].v, false, bfr[g][ni].v,
                        (short)0, acc[g][mi][ni], false, false);

        // ---- pack + store next-step tiles into the other buffer ----
        if (hasNext) {
#pragma unroll
            for (int i = 0; i < 4; ++i) {
                uint2 pa, pb;
                pa.x = pack2bf(ra[i].x, ra[i].y); pa.y = pack2bf(ra[i].z, ra[i].w);
                pb.x = pack2bf(rb[i].x, rb[i].y); pb.y = pack2bf(rb[i].z, rb[i].w);
                *(uint2*)&sA[nxt][ldsOff[i]] = pa;
                *(uint2*)&sB[nxt][ldsOff[i]] = pb;
            }
        }
        __syncthreads();
    }

    // ---- fused epilogue: bias + sigmoid gates + combine ----
#pragma unroll
    for (int mi = 0; mi < 2; ++mi)
#pragma unroll
        for (int ni = 0; ni < 2; ++ni) {
            const int col = nBase + cBase + ni * 16 + lrow;
            const float bi = bg[col];
            const float bf_ = bg[col + DD];
#pragma unroll
            for (int v = 0; v < 8; ++v) {
                const int row = mBase + rBase + mi * 16 + hb * 8 + v;
                const size_t idx = (size_t)row * DD + col;
                const float ig = acc[0][mi][ni][v] + bi;
                const float fg = acc[1][mi][ni][v] + bf_;
                out[idx] = sigm(ig) * x[idx] + sigm(fg) * avg[idx];
            }
        }
}

extern "C" void kernel_launch(void* const* d_in, const int* in_sizes, int n_in,
                              void* d_out, int out_size, void* d_ws, size_t ws_size,
                              hipStream_t stream) {
    const float* x  = (const float*)d_in[0];   // layer_in  (8,2048,2048) f32
    const float* Wg = (const float*)d_in[1];   // W_gate    (4096,4096)   f32
    const float* bg = (const float*)d_in[2];   // b_gate    (4096,)       f32

    float* out = (float*)d_out;                      // gating_out
    float* avg = out + (size_t)MTOT * DD;            // average_out (2nd output)
    float* part = out;  // scratch: overwritten by gated_gemm afterwards

    // 8*16*2048 = 262144 threads for each scan pass
    avg_partials<<<1024, 256, 0, stream>>>(x, part);
    avg_scan   <<<1024, 256, 0, stream>>>(x, part, avg);

    dim3 grid(DD / 64, MTOT / 128);                  // (32, 128), N fastest
    gated_gemm <<<grid, 256, 0, stream>>>(x, avg, Wg, bg, out);
}
